// Model_33921651704512
// MI455X (gfx1250) — matrix operand order repaired and checked by reading, not compiled, not run on hardware
//
#include <hip/hip_runtime.h>
#include <hip/hip_bf16.h>

// ---------------- types for WMMA ----------------
typedef __attribute__((ext_vector_type(16))) __bf16 v16bf;
typedef __attribute__((ext_vector_type(8)))  __bf16 v8bf;
typedef __attribute__((ext_vector_type(8)))  float  v8f;

// ---------------- bf16 helpers ----------------
__device__ __forceinline__ unsigned short f2bf(float f) {
    unsigned int u = __float_as_uint(f);
    u += 0x7FFFu + ((u >> 16) & 1u);   // round-to-nearest-even
    return (unsigned short)(u >> 16);
}
__device__ __forceinline__ float bf2f(unsigned short h) {
    return __uint_as_float(((unsigned int)h) << 16);
}

// ---------------- f32 -> bf16 weight conversion WITH transpose ----------------
// src: [K][N] f32 (pw weight, cin x cout)  ->  dst: [N][K] bf16 (Wt)
__global__ __launch_bounds__(256) void cvt_transpose_kernel(
    const float* __restrict__ src, unsigned short* __restrict__ dst, int K, int N) {
    int i = blockIdx.x * 256 + threadIdx.x;
    if (i >= K * N) return;
    int k = i % K;                  // dst coalesced: consecutive threads -> k
    int n = i / K;
    dst[i] = f2bf(src[(size_t)k * N + n]);
}

// ---------------- stem: 3x3 s2 conv 3->32 + bias + ReLU ----------------
__global__ __launch_bounds__(256) void stem_kernel(
    const float* __restrict__ x, const float* __restrict__ w,
    const float* __restrict__ b, unsigned short* __restrict__ out,
    int N, int H, int W, int Ho, int Wo) {
    int idx = blockIdx.x * 256 + threadIdx.x;
    int total = N * Ho * Wo * 32;
    if (idx >= total) return;
    int co = idx & 31;
    int t  = idx >> 5;
    int xo = t % Wo;
    int yo = (t / Wo) % Ho;
    int n  = t / (Wo * Ho);
    float acc = b[co];
#pragma unroll
    for (int ky = 0; ky < 3; ++ky) {
        int yi = yo * 2 - 1 + ky;
        if (yi < 0 || yi >= H) continue;
#pragma unroll
        for (int kx = 0; kx < 3; ++kx) {
            int xi = xo * 2 - 1 + kx;
            if (xi < 0 || xi >= W) continue;
            const float* px = x + ((size_t)(n * H + yi) * W + xi) * 3;
#pragma unroll
            for (int ci = 0; ci < 3; ++ci)
                acc += px[ci] * w[((ky * 3 + kx) * 3 + ci) * 32 + co];
        }
    }
    out[idx] = f2bf(fmaxf(acc, 0.0f));
}

// ---------------- depthwise 3x3 conv + bias + ReLU (bf16 in/out) ----------------
__global__ __launch_bounds__(256) void dw_kernel(
    const unsigned short* __restrict__ in, const float* __restrict__ w,
    const float* __restrict__ b, unsigned short* __restrict__ out,
    int N, int C, int H, int W, int Ho, int Wo, int s) {
    int idx = blockIdx.x * 256 + threadIdx.x;
    int total = N * Ho * Wo * C;
    if (idx >= total) return;
    int c  = idx % C;
    int t  = idx / C;
    int xo = t % Wo;
    int yo = (t / Wo) % Ho;
    int n  = t / (Wo * Ho);
    float acc = b[c];
#pragma unroll
    for (int ky = 0; ky < 3; ++ky) {
        int yi = yo * s - 1 + ky;
        if (yi < 0 || yi >= H) continue;
#pragma unroll
        for (int kx = 0; kx < 3; ++kx) {
            int xi = xo * s - 1 + kx;
            if (xi < 0 || xi >= W) continue;
            acc += bf2f(in[((size_t)(n * H + yi) * W + xi) * C + c]) *
                   w[(ky * 3 + kx) * C + c];
        }
    }
    out[idx] = f2bf(fmaxf(acc, 0.0f));
}

// ---------------- pointwise conv as WMMA GEMM: C = relu(A*Wt^T + bias) ----------------
// A:  [M][K] bf16 (activations, row-major)
// Wt: [N][K] bf16 (transposed weights, row-major) -> both tiles stage identically
// out bf16 [M][N]. Block tile 128x64, 8 waves (4x2 of 32x32 wave tiles), K step 32.
#define TM 128
#define TN 64
#define TK 32
__global__ __launch_bounds__(256) void pw_gemm_kernel(
    const unsigned short* __restrict__ A,
    const unsigned short* __restrict__ Wt,
    const float* __restrict__ bias,
    unsigned short* __restrict__ C,
    int M, int K, int N) {
    __shared__ unsigned short lA[TM][TK + 8];   // +8 pad keeps 16B align, spreads banks
    __shared__ unsigned short lB[TN][TK + 8];   // [ncol][k]

    const int tid   = threadIdx.x;
    const int lane  = tid & 31;
    const int wave  = tid >> 5;
    const int waveM = wave >> 1;     // 0..3
    const int waveN = wave & 1;      // 0..1
    const int blockM = blockIdx.x * TM;
    const int blockN = blockIdx.y * TN;

    v8f acc[2][2];
#pragma unroll
    for (int i = 0; i < 2; ++i)
#pragma unroll
        for (int j = 0; j < 2; ++j)
#pragma unroll
            for (int r = 0; r < 8; ++r) acc[i][j][r] = 0.0f;

    // global->LDS mapping (both operands: 16B vector copies, no scatter)
    const int arow  = tid >> 1;            // 0..127
    const int ahalf = (tid & 1) * 16;      // 0 or 16 (elements)
    const int bcol  = tid >> 2;            // 0..63
    const int bseg  = (tid & 3) * 8;       // 0,8,16,24 (elements)

    const unsigned short* aptr = A + (size_t)(blockM + arow) * K + ahalf;
    const unsigned short* bptr = Wt + (size_t)(blockN + bcol) * K + bseg;
    const bool arow_ok = (blockM + arow) < M;

    const int kTiles = K / TK;
    for (int kt = 0; kt < kTiles; ++kt) {
        // ---- stage A tile [128 rows][32 K] ----
        uint4 av0 = {0, 0, 0, 0}, av1 = {0, 0, 0, 0};
        if (arow_ok) {
            const uint4* p = (const uint4*)(aptr + kt * TK);
            av0 = p[0];
            av1 = p[1];
        }
        *(uint4*)&lA[arow][ahalf]     = av0;
        *(uint4*)&lA[arow][ahalf + 8] = av1;

        // ---- stage B tile [64 cols][32 K] (already transposed in global) ----
        *(uint4*)&lB[bcol][bseg] = *(const uint4*)(bptr + kt * TK);

        // prefetch next K-tile while this one computes (global_prefetch_b8)
        if (kt + 1 < kTiles) {
            if (arow_ok) __builtin_prefetch(aptr + (kt + 1) * TK, 0, 0);
            __builtin_prefetch(bptr + (kt + 1) * TK, 0, 0);
        }

        __syncthreads();

        // ---- fragment loads (ISA 7.12.2 lane layouts) ----
        const int koffA = (lane >> 4) * 8;    // A: lane<16 -> K 0-7/16-23; lane>=16 -> 8-15/24-31
        const int koffB = (lane >> 4) * 8;    // B: lane<16 -> K 0-7/16-23; lane>=16 -> 8-15/24-31
        v16bf af[2], bfr[2];
#pragma unroll
        for (int i = 0; i < 2; ++i) {
            const int mrow = waveM * 32 + i * 16 + (lane & 15);
            v8bf lo = *(const v8bf*)&lA[mrow][koffA];
            v8bf hi = *(const v8bf*)&lA[mrow][16 + koffA];
            af[i] = __builtin_shufflevector(lo, hi,
                     0, 1, 2, 3, 4, 5, 6, 7, 8, 9, 10, 11, 12, 13, 14, 15);
        }
#pragma unroll
        for (int j = 0; j < 2; ++j) {
            const int ncol = waveN * 32 + j * 16 + (lane & 15);
            v8bf lo = *(const v8bf*)&lB[ncol][koffB];
            v8bf hi = *(const v8bf*)&lB[ncol][16 + koffB];
            bfr[j] = __builtin_shufflevector(lo, hi,
                      0, 1, 2, 3, 4, 5, 6, 7, 8, 9, 10, 11, 12, 13, 14, 15);
        }

        // ---- 4 WMMAs per wave per K-step ----
#pragma unroll
        for (int i = 0; i < 2; ++i)
#pragma unroll
            for (int j = 0; j < 2; ++j)
                acc[i][j] = __builtin_amdgcn_wmma_f32_16x16x32_bf16(
                    false, af[i], false, bfr[j], (short)0, acc[i][j], false, false);

        __syncthreads();
    }

    // ---- epilogue: bias + ReLU + bf16 store ----
    const int rshift = (lane >> 4) << 3;  // C/D: lanes 16-31 hold rows +8
#pragma unroll
    for (int i = 0; i < 2; ++i) {
#pragma unroll
        for (int j = 0; j < 2; ++j) {
            const int colg = blockN + waveN * 32 + j * 16 + (lane & 15);
            const float bv = bias[colg];
#pragma unroll
            for (int r = 0; r < 8; ++r) {
                const int rowg = blockM + waveM * 32 + i * 16 + r + rshift;
                if (rowg < M) {
                    float v = acc[i][j][r] + bv;
                    C[(size_t)rowg * N + colg] = f2bf(fmaxf(v, 0.0f));
                }
            }
        }
    }
}

// ---------------- global average pool 7x7 -> [32][1024] f32 ----------------
__global__ __launch_bounds__(256) void pool_kernel(
    const unsigned short* __restrict__ in, float* __restrict__ out) {
    int idx = blockIdx.x * 256 + threadIdx.x;
    if (idx >= 32 * 1024) return;
    int c = idx & 1023;
    int n = idx >> 10;
    float s = 0.0f;
    const unsigned short* p = in + (size_t)n * 49 * 1024 + c;
#pragma unroll 7
    for (int i = 0; i < 49; ++i) s += bf2f(p[i * 1024]);
    out[idx] = s * (1.0f / 49.0f);
}

// ---------------- FC 1024 -> 1000 + bias ----------------
__global__ __launch_bounds__(256) void fc_kernel(
    const float* __restrict__ pooled, const float* __restrict__ w,
    const float* __restrict__ b, float* __restrict__ out) {
    int idx = blockIdx.x * 256 + threadIdx.x;
    if (idx >= 32 * 1000) return;
    int d = idx % 1000;
    int n = idx / 1000;
    float acc = b[d];
    const float* pr = pooled + n * 1024;
    for (int c = 0; c < 1024; ++c) acc += pr[c] * w[c * 1000 + d];
    out[idx] = acc;
}

// ---------------- host orchestration ----------------
extern "C" void kernel_launch(void* const* d_in, const int* in_sizes, int n_in,
                              void* d_out, int out_size, void* d_ws, size_t ws_size,
                              hipStream_t stream) {
    static const int cfg[13][3] = {
        {32, 64, 1},   {64, 128, 2},  {128, 128, 1}, {128, 256, 2},
        {256, 256, 1}, {256, 512, 2}, {512, 512, 1}, {512, 512, 1},
        {512, 512, 1}, {512, 512, 1}, {512, 512, 1}, {512, 1024, 2},
        {1024, 1024, 1}};

    const float* x       = (const float*)d_in[0];
    const float* conv1_w = (const float*)d_in[1];
    const float* conv1_b = (const float*)d_in[2];
    const float* dw_w[13];
    const float* dw_b[13];
    const float* pw_w[13];
    const float* pw_b[13];
    for (int i = 0; i < 13; ++i) {
        dw_w[i] = (const float*)d_in[3 + i];
        dw_b[i] = (const float*)d_in[16 + i];
        pw_w[i] = (const float*)d_in[29 + i];
        pw_b[i] = (const float*)d_in[42 + i];
    }
    const float* fc_w = (const float*)d_in[55];
    const float* fc_b = (const float*)d_in[56];

    // workspace layout (bf16 activations, ping buffers)
    const size_t ACT_ELEMS = 25690112;  // 32*112*112*64 (largest activation)
    unsigned short* buf0 = (unsigned short*)d_ws;            // block input/output
    unsigned short* buf1 = buf0 + ACT_ELEMS;                 // dw intermediate
    unsigned short* wbuf = buf1 + ACT_ELEMS;                 // bf16 Wt weights
    size_t wtot = 0;
    unsigned short* pwbf[13];
    for (int i = 0; i < 13; ++i) {
        pwbf[i] = wbuf + wtot;
        wtot += (size_t)cfg[i][0] * cfg[i][1];
    }
    float* pooled = (float*)(wbuf + ((wtot + 1) & ~(size_t)1));

    // 1) convert + transpose pointwise weights f32[K][N] -> bf16[N][K]
    for (int i = 0; i < 13; ++i) {
        int n = cfg[i][0] * cfg[i][1];
        cvt_transpose_kernel<<<(n + 255) / 256, 256, 0, stream>>>(
            pw_w[i], pwbf[i], cfg[i][0], cfg[i][1]);
    }

    // 2) stem conv -> buf0 (bf16, NHWC 32x112x112x32)
    {
        int total = 32 * 112 * 112 * 32;
        stem_kernel<<<(total + 255) / 256, 256, 0, stream>>>(
            x, conv1_w, conv1_b, buf0, 32, 224, 224, 112, 112);
    }

    // 3) 13 depthwise-separable blocks
    int Hcur = 112;
    for (int i = 0; i < 13; ++i) {
        const int cin = cfg[i][0], cout = cfg[i][1], s = cfg[i][2];
        const int Ho = Hcur / s;
        // depthwise: buf0 -> buf1
        int total = 32 * Ho * Ho * cin;
        dw_kernel<<<(total + 255) / 256, 256, 0, stream>>>(
            buf0, dw_w[i], dw_b[i], buf1, 32, cin, Hcur, Hcur, Ho, Ho, s);
        // pointwise WMMA GEMM: buf1 -> buf0
        const int M = 32 * Ho * Ho;
        dim3 grid((M + TM - 1) / TM, cout / TN);
        pw_gemm_kernel<<<grid, 256, 0, stream>>>(buf1, pwbf[i], pw_b[i], buf0,
                                                 M, cin, cout);
        Hcur = Ho;
    }

    // 4) global average pool -> pooled f32 [32][1024]
    pool_kernel<<<(32 * 1024 + 255) / 256, 256, 0, stream>>>(buf0, pooled);

    // 5) FC -> d_out f32 [32][1000]
    fc_kernel<<<(32 * 1000 + 255) / 256, 256, 0, stream>>>(
        pooled, fc_w, fc_b, (float*)d_out);
}
